// W4A4LinearQuarot_1176821039700
// MI455X (gfx1250) — 8-bit matrix operand order repaired and checked by reading, compiled, not run on hardware
//
#include <hip/hip_runtime.h>
#include <hip/hip_bf16.h>
#include <stdint.h>

typedef __attribute__((ext_vector_type(8))) int v8i;
typedef __attribute__((ext_vector_type(4))) int v4i;
typedef __attribute__((address_space(1))) v4i* gbl_v4i_ptr;
typedef __attribute__((address_space(3))) v4i* lds_v4i_ptr;

#define M_DIM 8192
#define K_DIM 4096
#define N_DIM 4096
#define CHUNK 256

#if __has_builtin(__builtin_amdgcn_global_load_async_to_lds_b128)
#define ASYNC_COPY_B128(gp, ldsp)                                              \
  __builtin_amdgcn_global_load_async_to_lds_b128(                              \
      (gbl_v4i_ptr)(gp), (lds_v4i_ptr)(ldsp), 0, 0)
#else
#define ASYNC_COPY_B128(gp, ldsp)                                              \
  asm volatile("global_load_async_to_lds_b128 %0, %1, off" ::                  \
               "v"((uint32_t)(uintptr_t)(                                      \
                   (__attribute__((address_space(3))) void*)(ldsp))),          \
               "v"(gp)                                                         \
               : "memory")
#endif

#if __has_builtin(__builtin_amdgcn_s_wait_asynccnt)
#define WAIT_ASYNC() __builtin_amdgcn_s_wait_asynccnt(0)
#else
#define WAIT_ASYNC() asm volatile("s_wait_asynccnt 0" ::: "memory")
#endif

__global__ __launch_bounds__(256) void w4a4_packw_kernel(
    const int* __restrict__ wqi, uint32_t* __restrict__ wq8, int total4) {
  int i = blockIdx.x * 256 + threadIdx.x;
  if (i >= total4) return;
  const int4 w = ((const int4*)wqi)[i];
  uint32_t packed = (uint32_t)(w.x & 0xFF)
                  | ((uint32_t)(w.y & 0xFF) << 8)
                  | ((uint32_t)(w.z & 0xFF) << 16)
                  | ((uint32_t)(w.w & 0xFF) << 24);
  wq8[i] = packed;
}

__global__ __launch_bounds__(256) void w4a4_quant_kernel(
    const float* __restrict__ x, signed char* __restrict__ xq8,
    float* __restrict__ xscale) {
  const int token = blockIdx.x;
  const int tid   = threadIdx.x;
  const int lane  = tid & 31;
  const int wv    = tid >> 5;

  const float* row = x + (size_t)token * K_DIM + (size_t)tid * 16;
  float vals[16];
  float mx = 0.0f;
#pragma unroll
  for (int i = 0; i < 16; ++i) {
    vals[i] = row[i];
    mx = fmaxf(mx, fabsf(vals[i]));
  }
#pragma unroll
  for (int off = 16; off >= 1; off >>= 1)
    mx = fmaxf(mx, __shfl_xor(mx, off, 32));

  __shared__ float smax[8];
  if (lane == 0) smax[wv] = mx;
  __syncthreads();
  if (wv == 0) {
    float mm = (lane < 8) ? smax[lane] : 0.0f;
#pragma unroll
    for (int off = 4; off >= 1; off >>= 1)
      mm = fmaxf(mm, __shfl_xor(mm, off, 32));
    if (lane == 0) smax[0] = mm;
  }
  __syncthreads();

  const float scale = fmaxf(smax[0] * (1.0f / 7.0f), 1e-5f);
  if (tid == 0) xscale[token] = scale;
  const float inv = 1.0f / scale;

  uint32_t packed[4];
#pragma unroll
  for (int g = 0; g < 4; ++g) {
    uint32_t p = 0;
#pragma unroll
    for (int i = 0; i < 4; ++i) {
      float r = rintf(vals[g * 4 + i] * inv);
      r = fminf(fmaxf(r, -8.0f), 7.0f);
      p |= ((uint32_t)((int)r & 0xFF)) << (8 * i);
    }
    packed[g] = p;
  }
  uint4 st = make_uint4(packed[0], packed[1], packed[2], packed[3]);
  *(uint4*)(xq8 + (size_t)token * K_DIM + (size_t)tid * 16) = st;
}

__global__ __launch_bounds__(256) void w4a4_gemm_kernel(
    const signed char* __restrict__ xq8, const signed char* __restrict__ wq8,
    const float* __restrict__ xs, const float* __restrict__ wscale,
    const float* __restrict__ bias, float* __restrict__ out) {
  __shared__ __align__(16) signed char sA[2][16 * CHUNK];

  const int tid  = threadIdx.x;
  const int lane = tid & 31;
  const int wib  = tid >> 5;

  const int mTile = blockIdx.x / 8;
  const int nGrp  = (blockIdx.x % 8) * 8 + wib;
  const int m0 = mTile * 16;
  const int n0 = nGrp * 64;

  const int hi  = lane >> 4;
  const int l16 = lane & 15;

  const int srow = tid >> 4;
  const int scol = (tid & 15) * 16;
  const signed char* gstage =
      xq8 + (size_t)(m0 + srow) * K_DIM + scol;
  signed char* lds0 = &sA[0][srow * CHUNK + scol];
  signed char* lds1 = &sA[1][srow * CHUNK + scol];

  const signed char* bRow0 = wq8 + (size_t)(n0 + l16) * K_DIM;

  v8i acc[4];
#pragma unroll
  for (int t = 0; t < 4; ++t) acc[t] = (v8i){0, 0, 0, 0, 0, 0, 0, 0};

  ASYNC_COPY_B128(gstage, lds0);
  WAIT_ASYNC();
  __syncthreads();

  const int nChunks = K_DIM / CHUNK;
  for (int c = 0; c < nChunks; ++c) {
    const int b = c & 1;
    if (c + 1 < nChunks) {
      const signed char* gnext = gstage + (size_t)(c + 1) * CHUNK;
      ASYNC_COPY_B128(gnext, (b == 0) ? lds1 : lds0);
    }

#pragma unroll
    for (int ks = 0; ks < 4; ++ks) {
      const int kc = ks * 64;
      const signed char* aRow = &sA[b][l16 * CHUNK + kc + hi * 16];
      uint2 p0 = *(const uint2*)(aRow);
      uint2 p1 = *(const uint2*)(aRow + 8);
      uint2 p2 = *(const uint2*)(aRow + 32);
      uint2 p3 = *(const uint2*)(aRow + 40);
      v8i a;
      a[0] = (int)p0.x; a[1] = (int)p0.y;
      a[2] = (int)p1.x; a[3] = (int)p1.y;
      a[4] = (int)p2.x; a[5] = (int)p2.y;
      a[6] = (int)p3.x; a[7] = (int)p3.y;

      const int kabs = c * CHUNK + kc;
#pragma unroll
      for (int t = 0; t < 4; ++t) {
        const signed char* bp =
            bRow0 + (size_t)t * 16 * K_DIM + kabs + hi * 16;
        uint4 q0 = *(const uint4*)(bp);
        uint4 q1 = *(const uint4*)(bp + 32);
        v8i bb;
        bb[0] = (int)q0.x; bb[1] = (int)q0.y; bb[2] = (int)q0.z; bb[3] = (int)q0.w;
        bb[4] = (int)q1.x; bb[5] = (int)q1.y; bb[6] = (int)q1.z; bb[7] = (int)q1.w;

        acc[t] = __builtin_amdgcn_wmma_i32_16x16x64_iu8(
            true, a, true, bb, acc[t], false, false);
      }
    }

    WAIT_ASYNC();
    __syncthreads();
  }

  float xsv[8];
#pragma unroll
  for (int j = 0; j < 8; ++j) xsv[j] = xs[m0 + hi * 8 + j];

#pragma unroll
  for (int t = 0; t < 4; ++t) {
    const int n = n0 + t * 16 + l16;
    const float wsn = wscale[n];
    const float bn  = bias[n];
#pragma unroll
    for (int j = 0; j < 8; ++j) {
      const int m = m0 + hi * 8 + j;
      out[(size_t)m * N_DIM + n] = (float)acc[t][j] * (xsv[j] * wsn) + bn;
    }
  }
}

extern "C" void kernel_launch(void* const* d_in, const int* in_sizes, int n_in,
                              void* d_out, int out_size, void* d_ws, size_t ws_size,
                              hipStream_t stream) {
  (void)in_sizes; (void)n_in; (void)out_size; (void)ws_size;

  const float* x        = (const float*)d_in[0];
  const int*   weight_q = (const int*)d_in[1];
  const float* wscale   = (const float*)d_in[2];
  const float* bias     = (const float*)d_in[3];
  float*       out      = (float*)d_out;

  signed char* wq8 = (signed char*)d_ws;
  signed char* xq8 = wq8 + (size_t)N_DIM * K_DIM;
  float*       xs  = (float*)(xq8 + (size_t)M_DIM * K_DIM);

  {
    const int total4 = (N_DIM * K_DIM) / 4;
    dim3 grid((total4 + 255) / 256), block(256);
    w4a4_packw_kernel<<<grid, block, 0, stream>>>(weight_q, (uint32_t*)wq8, total4);
  }
  {
    dim3 grid(M_DIM), block(256);
    w4a4_quant_kernel<<<grid, block, 0, stream>>>(x, xq8, xs);
  }
  {
    dim3 grid((M_DIM / 16) * 8), block(256);
    w4a4_gemm_kernel<<<grid, block, 0, stream>>>(xq8, wq8, xs, wscale, bias, out);
  }
}
